// NetworkODEModel_19404662243683
// MI455X (gfx1250) — matrix operand order repaired and checked by reading, not compiled, not run on hardware
//
#include <hip/hip_runtime.h>

typedef __attribute__((ext_vector_type(16))) _Float16 v16h;
typedef __attribute__((ext_vector_type(8)))  float    v8f;

#define BB   64
#define NN   128
#define HNH  64
#define HCH  64
#define VPAD 68   // 68*4B stride -> 16B-aligned rows, 4-bank lane stride (no conflicts)

// lrelu(x) == max(x, 0.01x): 2 VALU ops (mul+max), dual-issue friendly
__device__ __forceinline__ float lrelu(float v) { return fmaxf(v, 0.01f * v); }

// ---------------------------------------------------------------------------
// Node network: 8192 rows of 2 -> 64 -> 64 -> 1 (tiny; scalar VALU).
// Writes out[...,0] = x[...,1] and out[...,1] = node_out (coupling added later).
// ---------------------------------------------------------------------------
__global__ __launch_bounds__(256) void node_net_kernel(
    const float* __restrict__ x,   const float* __restrict__ Wn1,
    const float* __restrict__ bn1, const float* __restrict__ Wn2,
    const float* __restrict__ bn2, const float* __restrict__ Wno,
    const float* __restrict__ bno, float* __restrict__ out)
{
  __shared__ float sW2[HNH * HNH];
  for (int idx = threadIdx.x; idx < HNH * HNH; idx += 256) sW2[idx] = Wn2[idx];
  __syncthreads();

  const int gid = blockIdx.x * 256 + threadIdx.x;   // 0 .. B*N-1
  const float x0 = x[gid * 2 + 0];
  const float x1 = x[gid * 2 + 1];

  float h1[HNH];
#pragma unroll
  for (int h = 0; h < HNH; ++h)
    h1[h] = lrelu(fmaf(x0, Wn1[h], fmaf(x1, Wn1[HNH + h], bn1[h])));

  float o = bno[0];
  for (int n2 = 0; n2 < HNH; ++n2) {
    float t = bn2[n2];
#pragma unroll
    for (int k = 0; k < HNH; ++k) t = fmaf(h1[k], sW2[k * HNH + n2], t);
    o = fmaf(lrelu(t), Wno[n2], o);
  }
  out[gid * 2 + 0] = x1;
  out[gid * 2 + 1] = o;
}

// ---------------------------------------------------------------------------
// Coupling network via WMMA.
// Rows (b,i,j) form the GEMM M-dim; K = N = 64.  Layer-1 decomposed as
// c1 = lrelu(u_i + v_j + bc1) with u,v built in LDS (linearity of concat@Wc1).
// 256 threads = 8 waves per block; one wave per i, 8 i's share one sV stage.
// Per wave: 8 j-tiles x (2 K-steps x 4 N-tiles) v_wmma_f32_16x16x32_f16.
// ---------------------------------------------------------------------------
__global__ __launch_bounds__(256) void coupling_kernel(
    const float* __restrict__ x,   const float* __restrict__ Wc1,
    const float* __restrict__ bc1, const float* __restrict__ Wc2,
    const float* __restrict__ bc2, const float* __restrict__ Wco,
    const float* __restrict__ bco, const float* __restrict__ Ap,
    float* __restrict__ out)
{
  __shared__ float sV[NN * VPAD];  // v[b, j, h]                (34 KB, padded)
  __shared__ float sU[8 * HCH];    // u[b, i0..i0+7, h] + bc1   (2 KB)
  __shared__ float sA[8 * NN];     // sigmoid(Ap - I/eps) rows  (4 KB)

  const int tid = threadIdx.x;
  const int b   = blockIdx.x >> 4;         // 64 batches
  const int i0  = (blockIdx.x & 15) * 8;   // 8 i's per block (one per wave)

  { // stage v[b, j, :] : half a row per thread
    const int j  = tid & 127;
    const int h0 = (tid >> 7) * 32;
    const float xa = x[(b * NN + j) * 2 + 0];
    const float xb = x[(b * NN + j) * 2 + 1];
    for (int h = h0; h < h0 + 32; ++h)
      sV[j * VPAD + h] = fmaf(xa, Wc1[2 * HCH + h], xb * Wc1[3 * HCH + h]);
  }
  for (int idx = tid; idx < 8 * HCH; idx += 256) { // stage u + bc1
    const int w = idx >> 6, h = idx & 63;
    const int i = i0 + w;
    const float xa = x[(b * NN + i) * 2 + 0];
    const float xb = x[(b * NN + i) * 2 + 1];
    sU[idx] = fmaf(xa, Wc1[h], fmaf(xb, Wc1[HCH + h], bc1[h]));
  }
  for (int idx = tid; idx < 8 * NN; idx += 256) {  // stage adjacency rows
    const int w = idx >> 7, j = idx & 127;
    const int i = i0 + w;
    const float v = Ap[i * NN + j] - (i == j ? 1.0e5f : 0.0f);
    sA[idx] = 1.0f / (1.0f + __expf(-v));
  }
  __syncthreads();

  const int w    = tid >> 5;      // wave id -> i = i0 + w
  const int lane = tid & 31;
  const int hf   = lane >> 4;     // lane half (ISA A/B layout)
  const int jl   = lane & 15;

  // Preload Wc2 as f16 B-operands: 2 K-steps x 4 N-tiles (constant all kernel).
  // B 32x16 layout: element e of lane L holds Wc2[K = kt*32 + hf*8 + (e&7) + (e>>3)*16][N = nt*16 + jl].
  v16h Bop[2][4];
#pragma unroll
  for (int kt = 0; kt < 2; ++kt)
#pragma unroll
    for (int nt = 0; nt < 4; ++nt)
#pragma unroll
      for (int e = 0; e < 16; ++e)
        Bop[kt][nt][e] = (_Float16)Wc2[(kt * 32 + hf * 8 + (e & 7) + (e >> 3) * 16) * HCH + nt * 16 + jl];

  float bc2L[4], wcoL[4];
#pragma unroll
  for (int nt = 0; nt < 4; ++nt) {
    bc2L[nt] = bc2[nt * 16 + jl];
    wcoL[nt] = Wco[nt * 16 + jl];
  }
  const float bcov = bco[0];

  const float* U = sU + w * HCH;
  float coupAcc = 0.0f;

  for (int jt = 0; jt < 8; ++jt) {
    const int j0 = jt * 16;
    const float* V = sV + (j0 + jl) * VPAD;

    // A-operand 16x32 f16: lane jl = row M=j; element e<8 -> h0+e, e>=8 -> h0+16+(e-8).
    v16h a[2];
#pragma unroll
    for (int kt = 0; kt < 2; ++kt) {
      const int h0 = kt * 32 + hf * 8;
      float us[16], vs[16];
      *(float4*)(us + 0)  = *(const float4*)(U + h0);
      *(float4*)(us + 4)  = *(const float4*)(U + h0 + 4);
      *(float4*)(us + 8)  = *(const float4*)(U + h0 + 16);
      *(float4*)(us + 12) = *(const float4*)(U + h0 + 20);
      *(float4*)(vs + 0)  = *(const float4*)(V + h0);
      *(float4*)(vs + 4)  = *(const float4*)(V + h0 + 4);
      *(float4*)(vs + 8)  = *(const float4*)(V + h0 + 16);
      *(float4*)(vs + 12) = *(const float4*)(V + h0 + 20);
#pragma unroll
      for (int t = 0; t < 16; ++t) {
        const float s = us[t] + vs[t];
        a[kt][t] = (_Float16)fmaxf(s, 0.01f * s);
      }
    }

    float acc[8] = {0.f, 0.f, 0.f, 0.f, 0.f, 0.f, 0.f, 0.f};
#pragma unroll
    for (int nt = 0; nt < 4; ++nt) {
      v8f c = {};
      c = __builtin_amdgcn_wmma_f32_16x16x32_f16(false, a[0], false, Bop[0][nt],
                                                 (short)0, c, false, false);
      c = __builtin_amdgcn_wmma_f32_16x16x32_f16(false, a[1], false, Bop[1][nt],
                                                 (short)0, c, false, false);
#pragma unroll
      for (int r = 0; r < 8; ++r)                 // bc2 + lrelu + dot(Wco)
        acc[r] = fmaf(lrelu(c[r] + bc2L[nt]), wcoL[nt], acc[r]);
    }

    // Reduce over the 16 N-lanes (rows m = hf*8 + r -> j = j0 + m).
#pragma unroll
    for (int r = 0; r < 8; ++r) {
      acc[r] += __shfl_xor(acc[r], 1, 32);
      acc[r] += __shfl_xor(acc[r], 2, 32);
      acc[r] += __shfl_xor(acc[r], 4, 32);
      acc[r] += __shfl_xor(acc[r], 8, 32);
    }
    if (jl == 0) {
#pragma unroll
      for (int r = 0; r < 8; ++r)
        coupAcc = fmaf(sA[w * NN + j0 + hf * 8 + r], acc[r] + bcov, coupAcc);
    }
  }

  coupAcc += __shfl_xor(coupAcc, 16, 32);  // combine lane 0 + lane 16 halves
  if (lane == 0) {
    const int row = b * NN + i0 + w;
    out[row * 2 + 1] += coupAcc;           // node_out written by prior kernel
  }
}

// ---------------------------------------------------------------------------
extern "C" void kernel_launch(void* const* d_in, const int* in_sizes, int n_in,
                              void* d_out, int out_size, void* d_ws, size_t ws_size,
                              hipStream_t stream) {
  (void)in_sizes; (void)n_in; (void)d_ws; (void)ws_size; (void)out_size;
  const float* x   = (const float*)d_in[0];
  const float* Wn1 = (const float*)d_in[1];
  const float* bn1 = (const float*)d_in[2];
  const float* Wn2 = (const float*)d_in[3];
  const float* bn2 = (const float*)d_in[4];
  const float* Wno = (const float*)d_in[5];
  const float* bno = (const float*)d_in[6];
  const float* Wc1 = (const float*)d_in[7];
  const float* bc1 = (const float*)d_in[8];
  const float* Wc2 = (const float*)d_in[9];
  const float* bc2 = (const float*)d_in[10];
  const float* Wco = (const float*)d_in[11];
  const float* bco = (const float*)d_in[12];
  const float* Ap  = (const float*)d_in[13];
  float* out = (float*)d_out;

  node_net_kernel<<<BB * NN / 256, 256, 0, stream>>>(x, Wn1, bn1, Wn2, bn2, Wno, bno, out);
  coupling_kernel<<<BB * (NN / 8), 256, 0, stream>>>(x, Wc1, bc1, Wc2, bc2, Wco, bco, Ap, out);
}
